// Qalign_76175539962354
// MI455X (gfx1250) — compile-verified, matrix operand order checked by reading, not run on hardware
//
#include <hip/hip_runtime.h>
#include <hip/hip_fp16.h>

typedef __attribute__((ext_vector_type(16))) _Float16 v16h;
typedef __attribute__((ext_vector_type(8)))  _Float16 v8h;
typedef __attribute__((ext_vector_type(8)))  float    v8f;

#define N_PAIRS   8192
#define EMB_D     128
#define T_C       2.0f
#define ALPHA_C   4.0f
#define BETA_C    2.0f
#define EPS_C     1e-12f

#define COL_TILES   (N_PAIRS / 16)          // 512 column tiles
#define ROW_BLOCKS  (N_PAIRS / 256)         // 32: each block owns 16 row tiles
#define COL_CHUNKS  16
#define TPC         (COL_TILES / COL_CHUNKS) // 32 col tiles per block traversal

// ---------------------------------------------------------------- zero init
__global__ void Qalign_zero_kernel(float* __restrict__ rs_src,
                                   float* __restrict__ rs_pos,
                                   float* __restrict__ align_acc, int n) {
    int i = blockIdx.x * blockDim.x + threadIdx.x;
    if (i < n) { rs_src[i] = 0.0f; rs_pos[i] = 0.0f; }
    if (i == 0) *align_acc = 0.0f;
}

// ------------------------------------------- gather + L2 normalize + align
__global__ void __launch_bounds__(128)
Qalign_gather_norm_kernel(const float* __restrict__ emb,
                          const int*   __restrict__ u,
                          const int*   __restrict__ v,
                          _Float16*    __restrict__ src16,
                          _Float16*    __restrict__ pos16,
                          float*       __restrict__ align_acc) {
    const int i    = blockIdx.x;
    const int d    = threadIdx.x;          // 0..127
    const int lane = d & 31;
    const int wave = d >> 5;

    const float eu = emb[(size_t)u[i] * EMB_D + d];
    const float ev = emb[(size_t)v[i] * EMB_D + d];
    float su = eu * eu, sv = ev * ev, dp = eu * ev;

#pragma unroll
    for (int m = 16; m >= 1; m >>= 1) {     // wave32 reduction
        su += __shfl_xor(su, m, 32);
        sv += __shfl_xor(sv, m, 32);
        dp += __shfl_xor(dp, m, 32);
    }
    __shared__ float sU[4], sV[4], sD[4];
    if (lane == 0) { sU[wave] = su; sV[wave] = sv; sD[wave] = dp; }
    __syncthreads();
    su = sU[0] + sU[1] + sU[2] + sU[3];
    sv = sV[0] + sV[1] + sV[2] + sV[3];
    dp = sD[0] + sD[1] + sD[2] + sD[3];

    const float inu = 1.0f / fmaxf(sqrtf(su), EPS_C);
    const float inv = 1.0f / fmaxf(sqrtf(sv), EPS_C);
    src16[(size_t)i * EMB_D + d] = (_Float16)(eu * inu);
    pos16[(size_t)i * EMB_D + d] = (_Float16)(ev * inv);
    if (d == 0) atomicAdd(align_acc, T_C - T_C * (dp * inu * inv));
}

// ------------------------------------------------------- WMMA fragment load
// 16-bit A-matrix 16x32 layout (ISA 7.12.2): lane L holds row M=L%16;
// half L/16==0 -> K = 0..7 and 16..23; half L/16==1 -> K = 8..15 and 24..31.
template <typename PtrT>
__device__ __forceinline__ void load_frag4(PtrT r /* row base, 128 halfs */,
                                           int h, v16h frag[4]) {
#pragma unroll
    for (int f = 0; f < 4; ++f) {
        const int k0 = f * 32;
        v8h lo = *(const v8h*)(r + k0 + 8 * h);        // 16B aligned
        v8h hi = *(const v8h*)(r + k0 + 16 + 8 * h);
#pragma unroll
        for (int j = 0; j < 8; ++j) { frag[f][j] = lo[j]; frag[f][8 + j] = hi[j]; }
    }
}

// --------------------------- async stage one 16x128 f16 tile into LDS buffer
// 256 threads x 16B = 4KB. Low 32 bits of a flat LDS pointer are the
// wave-relative LDS byte offset expected in VDST (ISA 10.2 aperture rules).
__device__ __forceinline__ void stage_tile_async(const _Float16* __restrict__ e16,
                                                 int ct, _Float16* buf, int tid) {
    const int row = tid >> 4;              // 0..15
    const int seg = tid & 15;              // 16B segment within the row
    const _Float16* g = e16 + (size_t)(ct * 16 + row) * EMB_D + seg * 8;
    unsigned lds = (unsigned)(uintptr_t)(buf + row * EMB_D + seg * 8);
    asm volatile("global_load_async_to_lds_b128 %0, %1, off"
                 :: "v"(lds), "v"(g) : "memory");
}

// ------------------------------------- row sums of exp(T*<e_i,e_j> - T)
// grid = (32 row blocks, 16 column chunks), 8 waves per block.
// Each wave owns TWO 16-row tiles (M=32): B fragments read once from LDS
// feed 8 WMMAs, halving LDS traffic and barrier cost per FLOP.
// B tiles stream through triple-buffered LDS via async global->LDS DMA;
// one barrier per tile (stage(t+2) issued post-barrier, its buffer's last
// readers ran at iteration t-1).
__global__ void __launch_bounds__(256)
Qalign_uniform_rowsum_kernel(const _Float16* __restrict__ e16,
                             float*          __restrict__ rowsum) {
    __shared__ __align__(16) _Float16 sbuf[3][16 * EMB_D];   // 3 x 4KB

    const int tid   = threadIdx.x;
    const int wave  = tid >> 5;             // 0..7
    const int L     = tid & 31;             // lane
    const int h     = L >> 4;
    const int m     = L & 15;
    const int rt0   = blockIdx.x * 16 + wave * 2;   // first of two row tiles
    const int cbase = blockIdx.y * TPC;             // first column tile

    v16h afragA[4], afragB[4];
    load_frag4(e16 + (size_t)(rt0 * 16 + m) * EMB_D,       h, afragA);
    load_frag4(e16 + (size_t)((rt0 + 1) * 16 + m) * EMB_D, h, afragB);

    float accA[8], accB[8];
#pragma unroll
    for (int j = 0; j < 8; ++j) { accA[j] = 0.0f; accB[j] = 0.0f; }

    stage_tile_async(e16, cbase + 0, sbuf[0], tid);
    stage_tile_async(e16, cbase + 1, sbuf[1], tid);

    int cur = 0, nxt = 2;                   // t % 3 and (t+2) % 3, no division
    for (int t = 0; t < TPC; ++t) {
        // exactly one newer stage (tile t+1) outstanding here -> tile t done
        asm volatile("s_wait_asynccnt 0x1" ::: "memory");
        __syncthreads();                    // tile t visible; t-1 readers done

        if (t + 2 < TPC)
            stage_tile_async(e16, cbase + t + 2, sbuf[nxt], tid);

        v16h bfrag[4];
        load_frag4((const _Float16*)sbuf[cur] + (size_t)m * EMB_D, h, bfrag);

        // one accumulator chain per row tile: 8 WMMAs, ILP = 2
        v8f c0 = {}, c1 = {};
#pragma unroll
        for (int f = 0; f < 4; ++f) {
            c0 = __builtin_amdgcn_wmma_f32_16x16x32_f16(false, afragA[f], false,
                     bfrag[f], (short)0, c0, false, false);
            c1 = __builtin_amdgcn_wmma_f32_16x16x32_f16(false, afragB[f], false,
                     bfrag[f], (short)0, c1, false, false);
        }

#pragma unroll
        for (int j = 0; j < 8; ++j) {       // lane holds rows j + 8h
            accA[j] += __expf(T_C * c0[j] - T_C);
            accB[j] += __expf(T_C * c1[j] - T_C);
        }

        cur = (cur == 2) ? 0 : cur + 1;
        nxt = (nxt == 2) ? 0 : nxt + 1;
    }

    // sum over the 16 tile columns (lanes within each half), 32 atomics/wave
#pragma unroll
    for (int j = 0; j < 8; ++j) {
        float ra = accA[j], rb = accB[j];
        ra += __shfl_xor(ra, 8, 32);  rb += __shfl_xor(rb, 8, 32);
        ra += __shfl_xor(ra, 4, 32);  rb += __shfl_xor(rb, 4, 32);
        ra += __shfl_xor(ra, 2, 32);  rb += __shfl_xor(rb, 2, 32);
        ra += __shfl_xor(ra, 1, 32);  rb += __shfl_xor(rb, 1, 32);
        if (m == j) {
            atomicAdd(&rowsum[rt0 * 16 + j + 8 * h], ra);
            atomicAdd(&rowsum[(rt0 + 1) * 16 + j + 8 * h], rb);
        }
    }
}

// ---------------------------------------------------------------- finalize
__global__ void __launch_bounds__(256)
Qalign_finalize_kernel(const float* __restrict__ rs_src,
                       const float* __restrict__ rs_pos,
                       const float* __restrict__ align_acc,
                       float*       __restrict__ out) {
    __shared__ float sbuf[8];
    const int tid = threadIdx.x;
    float c = 0.0f;
    for (int i = tid; i < N_PAIRS; i += 256)
        c += logf(rs_src[i]) + logf(rs_pos[i]);
#pragma unroll
    for (int m = 16; m >= 1; m >>= 1) c += __shfl_xor(c, m, 32);
    if ((tid & 31) == 0) sbuf[tid >> 5] = c;
    __syncthreads();
    if (tid == 0) {
        float tot = 0.0f;
        for (int w = 0; w < 8; ++w) tot += sbuf[w];
        const float align = *align_acc / (float)N_PAIRS;
        out[0] = ALPHA_C * align + BETA_C * (tot / (float)N_PAIRS);
    }
}

// ---------------------------------------------------------------- launcher
extern "C" void kernel_launch(void* const* d_in, const int* in_sizes, int n_in,
                              void* d_out, int out_size, void* d_ws, size_t ws_size,
                              hipStream_t stream) {
    const int*   u   = (const int*)d_in[0];
    const int*   v   = (const int*)d_in[1];
    const float* emb = (const float*)d_in[2];

    char* ws = (char*)d_ws;
    _Float16* src16 = (_Float16*)ws;                                   // 2 MB
    _Float16* pos16 = (_Float16*)(ws + (size_t)N_PAIRS * EMB_D * 2);   // 2 MB
    float* rs_src   = (float*)(ws + (size_t)N_PAIRS * EMB_D * 4);      // 32 KB
    float* rs_pos   = rs_src + N_PAIRS;                                // 32 KB
    float* acc      = rs_pos + N_PAIRS;                                // 4 B

    Qalign_zero_kernel<<<(N_PAIRS + 255) / 256, 256, 0, stream>>>(
        rs_src, rs_pos, acc, N_PAIRS);

    Qalign_gather_norm_kernel<<<N_PAIRS, 128, 0, stream>>>(
        emb, u, v, src16, pos16, acc);

    dim3 grid(ROW_BLOCKS, COL_CHUNKS);
    Qalign_uniform_rowsum_kernel<<<grid, 256, 0, stream>>>(src16, rs_src);
    Qalign_uniform_rowsum_kernel<<<grid, 256, 0, stream>>>(pos16, rs_pos);

    Qalign_finalize_kernel<<<1, 256, 0, stream>>>(
        rs_src, rs_pos, acc, (float*)d_out);
}
